// Grapher_12309376270846
// MI455X (gfx1250) — matrix operand order repaired and checked by reading, not compiled, not run on hardware
//
#include <hip/hip_runtime.h>
#include <math.h>

// ---------------- constants ----------------
#define BB   8
#define CC   192
#define HH   56
#define NPIX 3136          // 56*56
#define KC   16            // clusters
#define KN   9             // neighbors
#define C2   384           // 2*C
#define NCAT 768           // concat projection width

// ---------------- WMMA types / helpers ----------------
typedef __attribute__((ext_vector_type(16))) __bf16         v16bf;
typedef __attribute__((ext_vector_type(16))) unsigned short v16u;
typedef __attribute__((ext_vector_type(8)))  float          v8f;

__device__ __forceinline__ unsigned short f32_to_bf16(float f) {
  unsigned int u = __builtin_bit_cast(unsigned int, f);
  unsigned int lsb = (u >> 16) & 1u;
  u += 0x7fffu + lsb;                       // round-to-nearest-even
  return (unsigned short)(u >> 16);
}

__device__ __forceinline__ float gelu_tanh(float x) {
  float x3 = x * x * x;
  return 0.5f * x * (1.0f + tanhf(0.7978845608028654f * (x + 0.044715f * x3)));
}

// A fragment: 16x32 bf16, row-major source, lda even -> two b128 loads per lane
__device__ __forceinline__ v16bf load_a_frag(const unsigned short* A, int lda,
                                             int m0, int k0, int lane) {
  int m  = m0 + (lane & 15);
  int kb = (lane >> 4) << 3;                // 0 or 8
  const unsigned int* p = (const unsigned int*)(A + (size_t)m * lda + k0);
  v16u t;
#pragma unroll
  for (int v = 0; v < 4; ++v) {
    unsigned int lo = p[(kb >> 1) + v];      // k = kb+2v, kb+2v+1
    unsigned int hi = p[8 + (kb >> 1) + v];  // k = 16+kb+2v, +1
    t[2*v]     = (unsigned short)(lo & 0xffffu);
    t[2*v + 1] = (unsigned short)(lo >> 16);
    t[8 + 2*v]     = (unsigned short)(hi & 0xffffu);
    t[8 + 2*v + 1] = (unsigned short)(hi >> 16);
  }
  return __builtin_bit_cast(v16bf, t);
}

// B fragment from pre-swizzled weights: [tile][lane][16] contiguous -> 2x b128
__device__ __forceinline__ v16bf load_b_frag_sw(const unsigned short* Bsw, int tilesN,
                                                int tk, int tn, int lane) {
  const v16u* p = (const v16u*)(Bsw + ((((size_t)tk * tilesN + tn) * 32 + lane) << 4));
  return __builtin_bit_cast(v16bf, *p);
}

__device__ __forceinline__ v8f wmma_bf16(v16bf a, v16bf b, v8f c) {
  return __builtin_amdgcn_wmma_f32_16x16x32_bf16(false, a, false, b, (short)0, c,
                                                 false, false);
}

// ---------------- K0: weight prep ----------------
__global__ void convert_bf16_kernel(const float* __restrict__ s,
                                    unsigned short* __restrict__ d, int n) {
  int i = blockIdx.x * blockDim.x + threadIdx.x;
  if (i < n) d[i] = f32_to_bf16(s[i]);
}

// wvcat[c][o] : o<384 -> Wtop-Wbot ; o>=384 -> Wbot   (nn_v_w is (384,384) row-major)
__global__ void wvcat_kernel(const float* __restrict__ wv,
                             unsigned short* __restrict__ outw) {
  int idx = blockIdx.x * blockDim.x + threadIdx.x;
  if (idx >= CC * NCAT) return;
  int c = idx / NCAT, o = idx % NCAT;
  float v = (o < C2) ? (wv[c * C2 + o] - wv[(c + CC) * C2 + o])
                     : wv[(c + CC) * C2 + (o - C2)];
  outw[c * NCAT + o] = f32_to_bf16(v);
}

// row-major bf16 (K x N) -> WMMA B-fragment swizzled [ktile*tilesN+ntile][lane][16]
__global__ void swizzle_b_kernel(const unsigned short* __restrict__ src,
                                 unsigned short* __restrict__ dst, int K, int N) {
  int idx = blockIdx.x * blockDim.x + threadIdx.x;
  if (idx >= K * N) return;
  int e    = idx & 15;
  int lane = (idx >> 4) & 31;
  int tile = idx >> 9;                       // 512 elems per 32x16 tile
  int tilesN = N >> 4;
  int tk = tile / tilesN, tn = tile % tilesN;
  int k = (tk << 5) + ((e >> 3) << 4) + ((lane >> 4) << 3) + (e & 7);
  int n = (tn << 4) + (lane & 15);
  dst[idx] = src[(size_t)k * N + n];
}

// ---------------- K1: stable counting sort per batch ----------------
__global__ void sort_kernel(const int* __restrict__ labels, int* __restrict__ order,
                            int* __restrict__ mapping, int* __restrict__ lsrt,
                            int* __restrict__ cstart) {
  int b = blockIdx.x * blockDim.x + threadIdx.x;
  if (b >= BB) return;
  const int* lab = labels + b * NPIX;
  int cnt[KC];
  for (int k = 0; k < KC; ++k) cnt[k] = 0;
  for (int i = 0; i < NPIX; ++i) cnt[lab[i] & (KC - 1)]++;
  int off[KC + 1];
  off[0] = 0;
  for (int k = 0; k < KC; ++k) off[k + 1] = off[k] + cnt[k];
  for (int k = 0; k <= KC; ++k) cstart[b * (KC + 1) + k] = off[k];
  int run[KC];
  for (int k = 0; k < KC; ++k) run[k] = off[k];
  for (int i = 0; i < NPIX; ++i) {
    int l = lab[i] & (KC - 1);
    int p = run[l]++;
    order[b * NPIX + p]   = i;
    mapping[b * NPIX + i] = p;
    lsrt[b * NPIX + p]    = l;
  }
}

// ---------------- K2: depthwise 7x7 conv + bias + residual -> (B,N,C) bf16 ----------------
__global__ void dwconv_kernel(const float* __restrict__ x, const float* __restrict__ w,
                              const float* __restrict__ bias,
                              unsigned short* __restrict__ xbt) {
  int bc = blockIdx.x;
  int b = bc / CC, c = bc % CC;
  __shared__ float tile[62 * 62];
  __shared__ float wk[49];
  const float* plane = x + ((size_t)b * CC + c) * NPIX;
  int tid = threadIdx.x;
  if (tid < 49) wk[tid] = w[c * 49 + tid];
  for (int i = tid; i < 62 * 62; i += 256) {
    int ph = i / 62 - 3, pw = i % 62 - 3;
    float v = 0.f;
    if (ph >= 0 && ph < HH && pw >= 0 && pw < HH) v = plane[ph * HH + pw];
    tile[i] = v;
  }
  __syncthreads();
  float bia = bias[c];
  for (int p = tid; p < NPIX; p += 256) {
    int h = p / HH, q = p % HH;
    float acc = 0.f;
#pragma unroll
    for (int kh = 0; kh < 7; ++kh)
#pragma unroll
      for (int kw = 0; kw < 7; ++kw)
        acc = fmaf(wk[kh * 7 + kw], tile[(h + kh) * 62 + (q + kw)], acc);
    float val = acc + bia + tile[(h + 3) * 62 + (q + 3)];
    xbt[((size_t)b * NPIX + p) * CC + c] = f32_to_bf16(val);
  }
}

// ---- WMMA GEMM, swizzled B, compile-time K/N, 16x32 output per wave ----
template <int K, int N>
__global__ void gemm_bf16_sw_kernel(const unsigned short* __restrict__ A,
                                    const unsigned short* __restrict__ Bsw,
                                    const float* __restrict__ bias,
                                    float* __restrict__ C, int M) {
  constexpr int tilesN = N >> 4;
  constexpr int groups = tilesN >> 1;        // two n-tiles per wave
  int wid  = (blockIdx.x * blockDim.x + threadIdx.x) >> 5;
  int lane = threadIdx.x & 31;
  int nGroups = (M >> 4) * groups;
  int nw = (gridDim.x * blockDim.x) >> 5;
  for (int t = wid; t < nGroups; t += nw) {
    int m0  = (t / groups) << 4;
    int tn0 = (t % groups) << 1;
    // prefetch next tile-group's A rows (global_prefetch_b8)
    int tnext = t + nw;
    if (tnext < nGroups) {
      const unsigned short* pf =
          A + (size_t)(((tnext / groups) << 4) + (lane & 15)) * K;
      __builtin_prefetch(pf, 0, 1);
    }
    v8f acc0 = {}, acc1 = {};
#pragma unroll
    for (int kk = 0; kk < K / 32; ++kk) {
      v16bf a  = load_a_frag(A, K, m0, kk << 5, lane);
      v16bf b0 = load_b_frag_sw(Bsw, tilesN, kk, tn0, lane);
      v16bf b1 = load_b_frag_sw(Bsw, tilesN, kk, tn0 + 1, lane);
      acc0 = wmma_bf16(a, b0, acc0);
      acc1 = wmma_bf16(a, b1, acc1);
    }
    int n0 = tn0 << 4;
    int n  = n0 + (lane & 15);
    int mb = m0 + ((lane >> 4) << 3);
    float bv0 = bias ? bias[n]      : 0.f;
    float bv1 = bias ? bias[n + 16] : 0.f;
#pragma unroll
    for (int e = 0; e < 8; ++e) {
      C[(size_t)(mb + e) * N + n]      = acc0[e] + bv0;
      C[(size_t)(mb + e) * N + n + 16] = acc1[e] + bv1;
    }
  }
}

// ---------------- K4: gather sorted rows + norms ----------------
__global__ void gather_kernel(const float* __restrict__ xf, const int* __restrict__ order,
                              float* __restrict__ xs, unsigned short* __restrict__ xsb,
                              float* __restrict__ invn, float* __restrict__ sqv) {
  int blk = blockIdx.x;
  int b = blk / NPIX, i = blk % NPIX;
  int tid = threadIdx.x;                    // 0..191 channel
  int src = order[b * NPIX + i];
  float v = xf[((size_t)b * NPIX + src) * CC + tid];
  size_t o = ((size_t)b * NPIX + i) * CC + tid;
  xs[o]  = v;
  xsb[o] = f32_to_bf16(v);
  float ss = v * v;
#pragma unroll
  for (int off = 16; off; off >>= 1) ss += __shfl_xor(ss, off, 32);
  __shared__ float wsum[6];
  if ((tid & 31) == 0) wsum[tid >> 5] = ss;
  __syncthreads();
  if (tid == 0) {
    float s = 0.f;
    for (int w = 0; w < 6; ++w) s += wsum[w];
    float nr  = sqrtf(s);
    float inv = 1.f / fmaxf(nr, 1e-12f);
    invn[b * NPIX + i] = inv;
    sqv[b * NPIX + i]  = s * inv * inv;     // ||xe||^2
  }
}

// ---------------- K5: centroids (deterministic, contiguous cluster ranges) ----------------
__global__ void centroid_kernel(const float* __restrict__ xs, const int* __restrict__ cstart,
                                float* __restrict__ cen) {
  int idx = blockIdx.x * blockDim.x + threadIdx.x;
  if (idx >= BB * KC * CC) return;
  int c = idx % CC, k = (idx / CC) % KC, b = idx / (CC * KC);
  int s = cstart[b * (KC + 1) + k], e = cstart[b * (KC + 1) + k + 1];
  float sum = 0.f;
  for (int j = s; j < e; ++j) sum += xs[((size_t)b * NPIX + j) * CC + c];
  float cnt = (float)(e - s);
  cen[idx] = sum / fmaxf(cnt, 1.f);
}

// ---------------- K6: same-cluster top-9 (one wave per sorted node) ----------------
__global__ void knn_kernel(const float* __restrict__ xs, const float* __restrict__ invn,
                           const float* __restrict__ sqv, const int* __restrict__ lsrt,
                           const int* __restrict__ cstart, int* __restrict__ nbr) {
  int gw   = (blockIdx.x * blockDim.x + threadIdx.x) >> 5;
  int lane = threadIdx.x & 31;
  int b = gw / NPIX, i = gw % NPIX;
  const float* xrow = xs + ((size_t)b * NPIX + i) * CC;
  float xi[6];
#pragma unroll
  for (int t = 0; t < 6; ++t) xi[t] = xrow[lane + 32 * t];
  float inv_i = invn[b * NPIX + i];
  float sq_i  = sqv[b * NPIX + i];
  int l = lsrt[b * NPIX + i];
  int s = cstart[b * (KC + 1) + l], e = cstart[b * (KC + 1) + l + 1];
  float bestd[KN];
  int   besti[KN];
#pragma unroll
  for (int k = 0; k < KN; ++k) { bestd[k] = __builtin_inff(); besti[k] = i; }
  for (int j = s; j < e; ++j) {
    const float* xj = xs + ((size_t)b * NPIX + j) * CC;
    float p = 0.f;
#pragma unroll
    for (int t = 0; t < 6; ++t) p = fmaf(xi[t], xj[lane + 32 * t], p);
#pragma unroll
    for (int off = 16; off; off >>= 1) p += __shfl_xor(p, off, 32);
    float d2 = sq_i + sqv[b * NPIX + j] - 2.f * inv_i * invn[b * NPIX + j] * p;
    // stable bubble-insert (all lanes identical, register-only)
    float cd = d2; int ci = j;
#pragma unroll
    for (int k = 0; k < KN; ++k) {
      bool take = cd < bestd[k];
      float td = bestd[k]; int ti = besti[k];
      if (take) { bestd[k] = cd; besti[k] = ci; cd = td; ci = ti; }
    }
  }
  if (lane == 0) {
#pragma unroll
    for (int k = 0; k < KN; ++k) nbr[((size_t)b * NPIX + i) * KN + k] = besti[k];
  }
}

// ---------------- K7b: edge combine gelu(av_i + bv_j + bias) -> max over 9 ----------------
__global__ void edge_combine_kernel(const float* __restrict__ avbv,
                                    const int* __restrict__ nbr,
                                    const float* __restrict__ nvb,
                                    float* __restrict__ hv) {
  int blk = blockIdx.x;
  int b = blk / NPIX, i = blk % NPIX;
  __shared__ int nb[KN];
  if (threadIdx.x < KN) nb[threadIdx.x] = nbr[((size_t)b * NPIX + i) * KN + threadIdx.x];
  __syncthreads();
  const float* av = avbv + ((size_t)b * NPIX + i) * NCAT;
  for (int c = threadIdx.x; c < C2; c += 128) {
    float a = av[c] + nvb[c];
    float m = -__builtin_inff();
#pragma unroll
    for (int k = 0; k < KN; ++k) {
      float bv = avbv[((size_t)b * NPIX + nb[k]) * NCAT + C2 + c];
      m = fmaxf(m, gelu_tanh(a + bv));
    }
    hv[((size_t)b * NPIX + i) * C2 + c] = m;
  }
}

// ---------------- K8: cluster projections (tiny, exact f32) ----------------
__global__ void cproj_kernel(const float* __restrict__ cen, const float* __restrict__ wc,
                             float* __restrict__ acbc) {
  int idx = blockIdx.x * blockDim.x + threadIdx.x;
  if (idx >= BB * KC * NCAT) return;
  int o = idx % NCAT, k = (idx / NCAT) % KC, b = idx / (NCAT * KC);
  const float* cr = cen + ((size_t)b * KC + k) * CC;
  float sum = 0.f;
  if (o < C2) {
    for (int c = 0; c < CC; ++c)
      sum = fmaf(cr[c], wc[c * C2 + o] - wc[(c + CC) * C2 + o], sum);
  } else {
    int oo = o - C2;
    for (int c = 0; c < CC; ++c) sum = fmaf(cr[c], wc[(c + CC) * C2 + oo], sum);
  }
  acbc[idx] = sum;
}

// ---------------- K9: cluster gelu-max ----------------
__global__ void hcmax_kernel(const float* __restrict__ acbc, const float* __restrict__ ncb,
                             float* __restrict__ hcm) {
  int idx = blockIdx.x * blockDim.x + threadIdx.x;
  if (idx >= BB * KC * C2) return;
  int c = idx % C2, k = (idx / C2) % KC, b = idx / (C2 * KC);
  float a = acbc[((size_t)b * KC + k) * NCAT + c] + ncb[c];
  float m = -__builtin_inff();
#pragma unroll
  for (int j = 0; j < KC; ++j)
    m = fmaxf(m, gelu_tanh(a + acbc[((size_t)b * KC + j) * NCAT + C2 + c]));
  hcm[idx] = m;
}

// ---------------- K10: fused un-sort + cluster-add + fc2 WMMA + residual ----------------
__global__ void final_gemm_kernel(const float* __restrict__ hv, const float* __restrict__ hcm,
                                  const int* __restrict__ mapping, const int* __restrict__ lsrt,
                                  const unsigned short* __restrict__ fc2sw,
                                  const float* __restrict__ fc2b,
                                  const float* __restrict__ x, float* __restrict__ out) {
  int blk = blockIdx.x;                       // B * (NPIX/16)
  int b  = blk / (NPIX / 16);
  int r0 = (blk % (NPIX / 16)) * 16;          // original node base row
  __shared__ unsigned short As[16 * C2];      // 12 KB bf16 A tile
  __shared__ int srow[16];
  int tid = threadIdx.x;                      // 128 threads = 4 waves
  if (tid < 16) srow[tid] = mapping[b * NPIX + r0 + tid];
  __syncthreads();
  for (int idx = tid; idx < 16 * C2; idx += 128) {
    int m = idx / C2, k = idx % C2;
    int s = srow[m];
    int lab = lsrt[b * NPIX + s];
    float vv = hv[((size_t)b * NPIX + s) * C2 + k] +
               hcm[((size_t)b * KC + lab) * C2 + k];
    As[idx] = f32_to_bf16(vv);
  }
  __syncthreads();
  int wid = tid >> 5, lane = tid & 31;
  for (int tn = wid; tn < CC / 16; tn += 4) { // 12 col tiles over 4 waves
    v8f acc = {};
#pragma unroll
    for (int kk = 0; kk < C2 / 32; ++kk) {    // fully unrolled: 12 WMMAs
      v16bf a = load_a_frag(&As[0], C2, 0, kk << 5, lane);
      v16bf w = load_b_frag_sw(fc2sw, CC / 16, kk, tn, lane);
      acc = wmma_bf16(a, w, acc);
    }
    int n = (tn << 4) + (lane & 15);
    float bb = fc2b[n];
    int mb = (lane >> 4) << 3;
#pragma unroll
    for (int e = 0; e < 8; ++e) {
      size_t o = ((size_t)b * CC + n) * NPIX + (r0 + mb + e);
      out[o] = acc[e] + bb + x[o];
    }
  }
}

// ---------------- host launcher ----------------
static inline char* wsp(char*& p, size_t bytes) {
  char* r = p;
  p += (bytes + 255) & ~(size_t)255;
  return r;
}

extern "C" void kernel_launch(void* const* d_in, const int* in_sizes, int n_in,
                              void* d_out, int out_size, void* d_ws, size_t ws_size,
                              hipStream_t stream) {
  const float* x      = (const float*)d_in[0];
  const int*   labels = (const int*)  d_in[1];
  const float* cpe_w  = (const float*)d_in[2];
  const float* cpe_b  = (const float*)d_in[3];
  const float* fc1_w  = (const float*)d_in[4];
  const float* fc1_b  = (const float*)d_in[5];
  const float* nn_v_w = (const float*)d_in[6];
  const float* nn_v_b = (const float*)d_in[7];
  const float* nn_c_w = (const float*)d_in[8];
  const float* nn_c_b = (const float*)d_in[9];
  const float* fc2_w  = (const float*)d_in[10];
  const float* fc2_b  = (const float*)d_in[11];
  float* out = (float*)d_out;

  char* p = (char*)d_ws;
  unsigned short* xbt   = (unsigned short*)wsp(p, (size_t)BB * NPIX * CC * 2);
  unsigned short* fc1wb = (unsigned short*)wsp(p, (size_t)CC * CC * 2);
  unsigned short* wvcat = (unsigned short*)wsp(p, (size_t)CC * NCAT * 2);
  unsigned short* fc2wb = (unsigned short*)wsp(p, (size_t)C2 * CC * 2);
  unsigned short* fc1sw = (unsigned short*)wsp(p, (size_t)CC * CC * 2);
  unsigned short* wvsw  = (unsigned short*)wsp(p, (size_t)CC * NCAT * 2);
  unsigned short* fc2sw = (unsigned short*)wsp(p, (size_t)C2 * CC * 2);
  float* xf    = (float*)wsp(p, (size_t)BB * NPIX * CC * 4);
  int*   order = (int*)  wsp(p, (size_t)BB * NPIX * 4);
  int*   mapng = (int*)  wsp(p, (size_t)BB * NPIX * 4);
  int*   lsrt  = (int*)  wsp(p, (size_t)BB * NPIX * 4);
  int*   cstrt = (int*)  wsp(p, (size_t)BB * (KC + 1) * 4);
  float* xs    = (float*)wsp(p, (size_t)BB * NPIX * CC * 4);
  unsigned short* xsb = (unsigned short*)wsp(p, (size_t)BB * NPIX * CC * 2);
  float* invn  = (float*)wsp(p, (size_t)BB * NPIX * 4);
  float* sqv   = (float*)wsp(p, (size_t)BB * NPIX * 4);
  float* cen   = (float*)wsp(p, (size_t)BB * KC * CC * 4);
  int*   nbr   = (int*)  wsp(p, (size_t)BB * NPIX * KN * 4);
  float* avbv  = (float*)wsp(p, (size_t)BB * NPIX * NCAT * 4);
  float* hv    = (float*)wsp(p, (size_t)BB * NPIX * C2 * 4);
  float* acbc  = (float*)wsp(p, (size_t)BB * KC * NCAT * 4);
  float* hcm   = (float*)wsp(p, (size_t)BB * KC * C2 * 4);

  // K0: weight prep: bf16 convert (+ algebraic split) then WMMA-fragment swizzle
  convert_bf16_kernel<<<(CC * CC + 255) / 256, 256, 0, stream>>>(fc1_w, fc1wb, CC * CC);
  wvcat_kernel<<<(CC * NCAT + 255) / 256, 256, 0, stream>>>(nn_v_w, wvcat);
  convert_bf16_kernel<<<(C2 * CC + 255) / 256, 256, 0, stream>>>(fc2_w, fc2wb, C2 * CC);
  swizzle_b_kernel<<<(CC * CC + 255) / 256, 256, 0, stream>>>(fc1wb, fc1sw, CC, CC);
  swizzle_b_kernel<<<(CC * NCAT + 255) / 256, 256, 0, stream>>>(wvcat, wvsw, CC, NCAT);
  swizzle_b_kernel<<<(C2 * CC + 255) / 256, 256, 0, stream>>>(fc2wb, fc2sw, C2, CC);

  // K1: stable counting sort per batch (deterministic)
  sort_kernel<<<1, 32, 0, stream>>>(labels, order, mapng, lsrt, cstrt);

  // K2: depthwise conv + bias + residual -> (B,N,C) bf16
  dwconv_kernel<<<BB * CC, 256, 0, stream>>>(x, cpe_w, cpe_b, xbt);

  // K3: fc1 GEMM  (25088x192)x(192x192) -> xf (B,N,C) f32
  gemm_bf16_sw_kernel<CC, CC><<<1024, 256, 0, stream>>>(xbt, fc1sw, fc1_b, xf, BB * NPIX);

  // K4: gather into sorted order + norms
  gather_kernel<<<BB * NPIX, CC, 0, stream>>>(xf, order, xs, xsb, invn, sqv);

  // K5: centroids
  centroid_kernel<<<(BB * KC * CC + 255) / 256, 256, 0, stream>>>(xs, cstrt, cen);

  // K6: same-cluster top-9 neighbors (wave per node)
  knn_kernel<<<(BB * NPIX) / 8, 256, 0, stream>>>(xs, invn, sqv, lsrt, cstrt, nbr);

  // K7a: factored edge-MLP projection  (25088x192)x(192x768) -> avbv
  gemm_bf16_sw_kernel<CC, NCAT><<<2048, 256, 0, stream>>>(xsb, wvsw, (const float*)nullptr,
                                                          avbv, BB * NPIX);

  // K7b: per-edge gelu + masked max (self-padded neighbor list)
  edge_combine_kernel<<<BB * NPIX, 128, 0, stream>>>(avbv, nbr, nn_v_b, hv);

  // K8/K9: cluster branch (tiny, f32 exact)
  cproj_kernel<<<(BB * KC * NCAT + 255) / 256, 256, 0, stream>>>(cen, nn_c_w, acbc);
  hcmax_kernel<<<(BB * KC * C2 + 255) / 256, 256, 0, stream>>>(acbc, nn_c_b, hcm);

  // K10: fused un-sort + cluster-add + fc2 WMMA + bias + residual -> NCHW out
  final_gemm_kernel<<<BB * (NPIX / 16), 128, 0, stream>>>(hv, hcm, mapng, lsrt,
                                                          fc2sw, fc2_b, x, out);

  (void)in_sizes; (void)n_in; (void)out_size; (void)ws_size;
}
